// RPN_65970697666754
// MI455X (gfx1250) — matrix operand order repaired and checked by reading, not compiled, not run on hardware
//
#include <hip/hip_runtime.h>
#include <hip/hip_bf16.h>

typedef __attribute__((ext_vector_type(16))) _Float16 v16h;
typedef __attribute__((ext_vector_type(8)))  _Float16 v8h;
typedef __attribute__((ext_vector_type(8)))  float    v8f;

#define BB 8
#define CC 256
#define HH 128
#define WW 128
#define PH 130                   /* padded H/W for halo-free conv */
#define AAN 15
#define HWA (HH*WW*AAN)          /* 245760 */
#define PRE 2000
#define POST 1000
#define NMSW 63                  /* ceil(2000/32) */
#define SCALE_CLAMP 4.135166556742356f
#define NMS_THRESH 0.7f

__device__ __forceinline__ unsigned fkey(float f) {
  unsigned u = __float_as_uint(f);
  return (u & 0x80000000u) ? ~u : (u | 0x80000000u);
}
__device__ __forceinline__ float unfkey(unsigned k) {
  unsigned u = (k & 0x80000000u) ? (k & 0x7fffffffu) : ~k;
  return __uint_as_float(u);
}
__device__ __forceinline__ float neg_inf() { return __int_as_float(0xff800000); }

#define WMMA_F16(Am, Bm, Cm) \
  __builtin_amdgcn_wmma_f32_16x16x32_f16(false, (Am), false, (Bm), (short)0, (Cm), false, false)

// ---------------- layout conversion / weight pre-swizzle ----------------

// zero the 1-pixel halo ring of the padded NHWC-f16 feature tensor
__global__ void k_border(_Float16* __restrict__ dst) {
  int g = blockIdx.x * 256 + threadIdx.x;          // 8 * 516 * 256 = 1,056,768
  if (g >= BB * 516 * 256) return;
  int c = g & 255;
  int p = (g >> 8) % 516;
  int b = g / (516 * 256);
  int y, x;
  if (p < 130)      { y = 0;   x = p; }
  else if (p < 260) { y = 129; x = p - 130; }
  else { int q = p - 260; y = 1 + (q >> 1); x = (q & 1) ? 129 : 0; }
  dst[((size_t)(b * PH + y) * PH + x) * 256 + c] = (_Float16)0.f;
}

// NCHW f32 -> padded NHWC f16 via LDS transpose (both sides coalesced)
// tile: 32 channels x 64 pixels (one row segment)
__global__ void __launch_bounds__(256) k_cvt_feat(const float* __restrict__ src,
                                                  _Float16* __restrict__ dst) {
  __shared__ _Float16 lds[32][66];
  int blk = blockIdx.x;                            // 8 * 128 * 8 * 2 = 16384
  int xseg = blk & 1;
  int cseg = (blk >> 1) & 7;
  int y = (blk >> 4) & 127;
  int b = blk >> 11;
  int t = threadIdx.x;
  int x0 = xseg * 64, c0 = cseg * 32;

  int px = t & 63, cg = t >> 6;                    // read: px fastest -> coalesced f32
#pragma unroll
  for (int k = 0; k < 8; ++k) {
    int c = cg * 8 + k;
    lds[c][px] = (_Float16)src[((size_t)(b * CC + c0 + c) * HH + y) * WW + x0 + px];
  }
  __syncthreads();
  int cw = t & 31, pw = t >> 5;                    // write: c fastest -> coalesced f16
  size_t dbase = ((size_t)(b * PH + y + 1) * PH + x0 + 1) * 256 + c0 + cw;
#pragma unroll
  for (int k = 0; k < 8; ++k) {
    int px2 = pw * 8 + k;
    dst[dbase + (size_t)px2 * 256] = lds[cw][px2];
  }
}

// conv_w (OIHW 256x256x3x3) -> B-fragment layout [q=72][coTile=16][lane=32][16 halves]
__global__ void k_pack_w3(const float* __restrict__ w, _Float16* __restrict__ dst) {
  int o = blockIdx.x * 256 + threadIdx.x;          // 589,824
  int j = o & 15, lane = (o >> 4) & 31, ct = (o >> 9) & 15, q = o >> 13;
  int ky = q / 24, kx = (q >> 3) % 3, ci = (q & 7) * 32 + ((lane < 16) ? 0 : 8) + (j < 8 ? j : j + 8);
  int co = ct * 16 + (lane & 15);
  dst[o] = (_Float16)w[((co * CC + ci) * 3 + ky) * 3 + kx];
}

// obj_w(15x256) + delta_w(60x256) -> padded 80 co, [q=8][coTile=5][lane=32][16]
__global__ void k_pack_w1(const float* __restrict__ wo, const float* __restrict__ wd,
                          _Float16* __restrict__ dst) {
  int o = blockIdx.x * 256 + threadIdx.x;          // 20,480
  if (o >= 8 * 5 * 32 * 16) return;
  int j = o & 15, lane = (o >> 4) & 31, ct = (o >> 9) % 5, q = o / 2560;
  int ci = q * 32 + ((lane < 16) ? 0 : 8) + (j < 8 ? j : j + 8);
  int co = ct * 16 + (lane & 15);
  float v = 0.f;
  if (co < 15) v = wo[co * CC + ci];
  else if (co < 75) v = wd[(co - 15) * CC + ci];
  dst[o] = (_Float16)v;
}

// ---------------- 3x3 conv as implicit GEMM (WMMA f16) ----------------
// Wave tile: 32 pixels x 32 out-channels (4 accumulators, A/B fragments reused 2x each).
// Features padded -> branch-free inner loop; 3x3 taps fully unrolled -> constant offsets.
__global__ void __launch_bounds__(256) k_conv3x3(const _Float16* __restrict__ feat,
                                                 const _Float16* __restrict__ wswz,
                                                 const float* __restrict__ bias,
                                                 _Float16* __restrict__ out) {
  int wv = blockIdx.x * 8 + (threadIdx.x >> 5);    // 32768 waves total
  int lane = threadIdx.x & 31;
  int laneLo = lane & 15;
  int hi = lane >> 4;
  int ct2 = wv & 7;                                // 32-co tile 0..7
  int pt = wv >> 3;                                // 32-px tile 0..4095
  int p0 = pt * 32;
  int b = p0 >> 14, rem = p0 & 16383, y = rem >> 7, x0 = rem & 127;

  v8f a00 = {}, a01 = {}, a10 = {}, a11 = {};
  // ISA 16-bit A 16x32 layout: lane<16 holds K 0..7 & 16..23, lane>=16 holds 8..15 & 24..31
  const _Float16* aBase = feat + (((size_t)(b * PH + y)) * PH + x0 + laneLo) * 256 + (hi ? 8 : 0);
  const _Float16* bBase = wswz + ((size_t)(ct2 * 2) * 32 + lane) * 16;

#pragma unroll
  for (int ky = 0; ky < 3; ++ky) {
#pragma unroll
    for (int kx = 0; kx < 3; ++kx) {
      const _Float16* ap = aBase + ((size_t)ky * PH + kx) * 256;
      const _Float16* bp = bBase + (size_t)((ky * 3 + kx) * 8) * 8192;
#pragma unroll 2
      for (int c = 0; c < 8; ++c) {
        union { v16h v; v8h h[2]; } A0, A1, B0, B1;
        const _Float16* a0p = ap + c * 32;
        A0.h[0] = *(const v8h*)a0p;
        A0.h[1] = *(const v8h*)(a0p + 16);
        A1.h[0] = *(const v8h*)(a0p + 4096);       // pixels +16 -> +16*256 halves
        A1.h[1] = *(const v8h*)(a0p + 4096 + 16);
        const _Float16* b0p = bp + (size_t)c * 8192;
        B0.h[0] = *(const v8h*)b0p;
        B0.h[1] = *(const v8h*)(b0p + 8);
        B1.h[0] = *(const v8h*)(b0p + 512);        // co tile +16 -> +32*16 halves
        B1.h[1] = *(const v8h*)(b0p + 512 + 8);
        a00 = WMMA_F16(A0.v, B0.v, a00);
        a01 = WMMA_F16(A0.v, B1.v, a01);
        a10 = WMMA_F16(A1.v, B0.v, a10);
        a11 = WMMA_F16(A1.v, B1.v, a11);
      }
    }
  }

  int co = ct2 * 32 + laneLo;
  float bs0 = bias[co], bs1 = bias[co + 16];
  size_t obase = ((size_t)(b * HH + y) * WW + x0) * 256 + co;
#pragma unroll
  for (int v = 0; v < 8; ++v) {                    // D: row M = v + hi*8, col N = laneLo
    int r0 = v + hi * 8;
    float r;
    r = a00[v] + bs0; r = r > 0.f ? r : 0.f;
    out[obase + (size_t)r0 * 256] = (_Float16)r;
    r = a01[v] + bs1; r = r > 0.f ? r : 0.f;
    out[obase + (size_t)r0 * 256 + 16] = (_Float16)r;
    r = a10[v] + bs0; r = r > 0.f ? r : 0.f;
    out[obase + (size_t)(r0 + 16) * 256] = (_Float16)r;
    r = a11[v] + bs1; r = r > 0.f ? r : 0.f;
    out[obase + (size_t)(r0 + 16) * 256 + 16] = (_Float16)r;
  }
}

// ---------------- 1x1 convs (obj + deltas) as WMMA GEMM ----------------
__global__ void __launch_bounds__(256) k_conv1x1(const _Float16* __restrict__ t,
                                                 const _Float16* __restrict__ w1,
                                                 const float* __restrict__ objb,
                                                 const float* __restrict__ delb,
                                                 float* __restrict__ scores,
                                                 float* __restrict__ deltas) {
  int wv = blockIdx.x * 8 + (threadIdx.x >> 5);
  if (wv >= 8192 * 5) return;
  int lane = threadIdx.x & 31, laneLo = lane & 15, hi = lane >> 4;
  int ct = wv % 5, pt = wv / 5;
  int p0 = pt * 16, b = p0 >> 14, rem = p0 & 16383, y = rem >> 7, x0 = rem & 127;
  int xm = x0 + laneLo;

  v8f acc = {};
  const _Float16* abase = t + (((size_t)(b * HH + y) * WW + xm) << 8) + (hi ? 8 : 0);
  const _Float16* wbase = w1 + ((size_t)ct * 32 + lane) * 16;

#pragma unroll
  for (int q = 0; q < 8; ++q) {
    union { v16h v; v8h h[2]; } A, Bm;
    const _Float16* ap = abase + q * 32;
    A.h[0] = *(const v8h*)ap;
    A.h[1] = *(const v8h*)(ap + 16);
    const _Float16* bp = wbase + (size_t)q * 2560; // [q][ct(5)][lane][16]
    Bm.h[0] = *(const v8h*)bp;
    Bm.h[1] = *(const v8h*)(bp + 8);
    acc = WMMA_F16(A.v, Bm.v, acc);
  }

  int co = ct * 16 + laneLo;                       // 0..79, 75..79 padded
#pragma unroll
  for (int v = 0; v < 8; ++v) {
    int xr = x0 + v + hi * 8;
    int pix = y * WW + xr;
    float val = acc[v];
    if (co < 15) {
      scores[(size_t)b * HWA + pix * AAN + co] = val + objb[co];
    } else if (co < 75) {
      int c2 = co - 15, a = c2 >> 2, d = c2 & 3;
      deltas[((size_t)b * HWA + pix * AAN + a) * 4 + d] = val + delb[c2];
    }
  }
}

// ---------------- box decode (in place: deltas -> boxes) ----------------
__global__ void k_decode(float* __restrict__ bx) {
  int g = blockIdx.x * 256 + threadIdx.x;
  if (g >= BB * HWA) return;
  int i = g % HWA;
  int a = i % AAN, pix = i / AAN, x = pix % WW, y = pix / WW;
  float gx = x * 16.f, gy = y * 16.f;
  int s = a / 3, r = a % 3;
  float size = (float)(32 << s);
  float ratio = 0.5f * (float)(1 << r);
  float w = sqrtf(size * size / ratio), h = w * ratio;
  float* p = bx + (size_t)g * 4;
  float dx = p[0], dy = p[1], dw = p[2], dh = p[3];
  dw = fminf(dw, SCALE_CLAMP); dh = fminf(dh, SCALE_CLAMP);
  float px = dx * w + gx, py = dy * h + gy;
  float pw = __expf(dw) * w, ph = __expf(dh) * h;
  p[0] = px - 0.5f * pw; p[1] = py - 0.5f * ph;
  p[2] = px + 0.5f * pw; p[3] = py + 0.5f * ph;
}

// ---------------- top-K selection ----------------
__global__ void k_zero(unsigned* __restrict__ p, int n) {
  int g = blockIdx.x * 256 + threadIdx.x;
  if (g < n) p[g] = 0;
}

__global__ void k_hist(const float* __restrict__ scores, unsigned* __restrict__ hist) {
  __shared__ unsigned lh[2048];
  int b = blockIdx.x / 96, seg = blockIdx.x % 96;
  for (int t = threadIdx.x; t < 2048; t += 256) lh[t] = 0;
  __syncthreads();
  const float* sp = scores + (size_t)b * HWA + seg * 2560;
#pragma unroll
  for (int it = 0; it < 10; ++it)
    atomicAdd(&lh[fkey(sp[it * 256 + threadIdx.x]) >> 21], 1u);
  __syncthreads();
  for (int t = threadIdx.x; t < 2048; t += 256)
    if (lh[t]) atomicAdd(&hist[b * 2048 + t], lh[t]);
}

__global__ void k_select(const unsigned* __restrict__ hist, unsigned* __restrict__ tc) {
  int b = blockIdx.x;
  if (threadIdx.x == 0) {
    unsigned cum = 0, hi = 0; int T = 0;
    for (int t = 2047; t >= 0; --t) {
      unsigned h = hist[b * 2048 + t];
      if (cum + h >= PRE) { T = t; hi = cum; break; }
      cum += h;
    }
    tc[b * 4] = (unsigned)T; tc[b * 4 + 1] = hi; tc[b * 4 + 2] = 0; tc[b * 4 + 3] = 0;
  }
}

__global__ void k_compact(const float* __restrict__ scores, unsigned* __restrict__ tc,
                          unsigned* __restrict__ skey, int* __restrict__ sidx) {
  int g = blockIdx.x * 256 + threadIdx.x;
  if (g >= BB * HWA) return;
  int b = g / HWA, i = g % HWA;
  unsigned key = fkey(scores[g]);
  unsigned T = tc[b * 4], hi = tc[b * 4 + 1];
  unsigned bk = key >> 21;
  if (bk > T) {
    unsigned pos = atomicAdd(&tc[b * 4 + 2], 1u);
    skey[b * 2048 + pos] = key; sidx[b * 2048 + pos] = i;
  } else if (bk == T) {
    unsigned pos = hi + atomicAdd(&tc[b * 4 + 3], 1u);
    if (pos < PRE) { skey[b * 2048 + pos] = key; sidx[b * 2048 + pos] = i; }
  }
}

// bitonic sort 2048 (2000 real + 48 zero-key pad) desc, then gather + clip
__global__ void __launch_bounds__(1024) k_sortgather(const unsigned* __restrict__ skey,
                                                     const int* __restrict__ sidx,
                                                     const float* __restrict__ boxes,
                                                     float* __restrict__ sb,
                                                     float* __restrict__ ss) {
  __shared__ unsigned kk[2048];
  __shared__ int ix[2048];
  int b = blockIdx.x, tid = threadIdx.x;
  for (int e = tid; e < 2048; e += 1024) { kk[e] = skey[b * 2048 + e]; ix[e] = sidx[b * 2048 + e]; }
  __syncthreads();
  for (int k = 2; k <= 2048; k <<= 1)
    for (int j = k >> 1; j > 0; j >>= 1) {
      for (int e = tid; e < 2048; e += 1024) {
        int p = e ^ j;
        if (p > e) {
          bool desc = ((e & k) == 0);
          unsigned a = kk[e], c = kk[p];
          if (desc ? (a < c) : (a > c)) {
            kk[e] = c; kk[p] = a;
            int t = ix[e]; ix[e] = ix[p]; ix[p] = t;
          }
        }
      }
      __syncthreads();
    }
  for (int e = tid; e < PRE; e += 1024) {
    float sc = unfkey(kk[e]);
    const float* bp = boxes + ((size_t)b * HWA + ix[e]) * 4;
    float x1 = fminf(fmaxf(bp[0], 0.f), 2048.f);
    float y1 = fminf(fmaxf(bp[1], 0.f), 2048.f);
    float x2 = fminf(fmaxf(bp[2], 0.f), 2048.f);
    float y2 = fminf(fmaxf(bp[3], 0.f), 2048.f);
    float* o = sb + ((size_t)b * PRE + e) * 4;
    o[0] = x1; o[1] = y1; o[2] = x2; o[3] = y2;
    ss[b * PRE + e] = sc;
  }
}

// ---------------- NMS ----------------
__global__ void k_nmsmask(const float* __restrict__ sb, unsigned* __restrict__ mask) {
  int g = blockIdx.x * 256 + threadIdx.x;
  if (g >= BB * PRE * NMSW) return;
  int w = g % NMSW, i = (g / NMSW) % PRE, b = g / (NMSW * PRE);
  const float* bi = sb + ((size_t)b * PRE + i) * 4;
  float x1 = bi[0], y1 = bi[1], x2 = bi[2], y2 = bi[3];
  float ai = (x2 - x1) * (y2 - y1);
  unsigned m = 0;
  int j0 = w * 32;
  for (int k = 0; k < 32; ++k) {
    int j = j0 + k;
    if (j <= i || j >= PRE) continue;
    const float* bj = sb + ((size_t)b * PRE + j) * 4;
    float xx1 = fmaxf(x1, bj[0]), yy1 = fmaxf(y1, bj[1]);
    float xx2 = fminf(x2, bj[2]), yy2 = fminf(y2, bj[3]);
    float inter = fmaxf(xx2 - xx1, 0.f) * fmaxf(yy2 - yy1, 0.f);
    float aj = (bj[2] - bj[0]) * (bj[3] - bj[1]);
    float iou = inter / (ai + aj - inter + 1e-9f);
    if (iou > NMS_THRESH) m |= (1u << k);
  }
  mask[g] = m;
}

__global__ void __launch_bounds__(256) k_nmsfinal(const unsigned* __restrict__ mask,
                                                  const float* __restrict__ sb,
                                                  const float* __restrict__ ss,
                                                  float* __restrict__ out) {
  __shared__ unsigned remv[NMSW];
  __shared__ int kept[POST], supp[POST];
  __shared__ int nk, ns;
  int b = blockIdx.x, tid = threadIdx.x;
  if (tid < NMSW) remv[tid] = 0;
  if (tid == 0) { nk = 0; ns = 0; }
  __syncthreads();
  for (int i = 0; i < PRE; ++i) {
    unsigned sup = (remv[i >> 5] >> (i & 31)) & 1u;
    __syncthreads();
    if (!sup) {
      if (tid < NMSW) remv[tid] |= mask[((size_t)b * PRE + i) * NMSW + tid];
      if (tid == 0) { if (nk < POST) kept[nk] = i; nk = nk + 1; }
    } else {
      if (tid == 0) { if (ns < POST) supp[ns] = i; ns = ns + 1; }
    }
    __syncthreads();
  }
  int nkc = nk < POST ? nk : POST;
  for (int j = tid; j < POST; j += 256) {
    int src; float sc;
    if (j < nkc) { src = kept[j]; sc = ss[b * PRE + src]; }
    else { int r = j - nkc; src = (r < POST) ? supp[r] : 0; sc = neg_inf(); }
    const float* bp = sb + ((size_t)b * PRE + src) * 4;
    float* o = out + ((size_t)b * POST + j) * 5;
    o[0] = bp[0]; o[1] = bp[1]; o[2] = bp[2]; o[3] = bp[3]; o[4] = sc;
  }
}

// ---------------- host launcher ----------------
extern "C" void kernel_launch(void* const* d_in, const int* in_sizes, int n_in,
                              void* d_out, int out_size, void* d_ws, size_t ws_size,
                              hipStream_t stream) {
  const float* features = (const float*)d_in[0];
  const float* conv_w   = (const float*)d_in[1];
  const float* conv_b   = (const float*)d_in[2];
  const float* obj_w    = (const float*)d_in[3];
  const float* obj_b    = (const float*)d_in[4];
  const float* delta_w  = (const float*)d_in[5];
  const float* delta_b  = (const float*)d_in[6];
  float* out = (float*)d_out;

  char* ws = (char*)d_ws;
  size_t off = 0;
  auto alloc = [&](size_t bytes) { size_t o = off; off = (off + bytes + 255) & ~(size_t)255; return o; };

  _Float16* featH  = (_Float16*)(ws + alloc((size_t)BB * PH * PH * CC * 2));  // 69 MB padded
  _Float16* wswz   = (_Float16*)(ws + alloc((size_t)2304 * 256 * 2));
  _Float16* w1swz  = (_Float16*)(ws + alloc((size_t)8 * 5 * 32 * 16 * 2));
  _Float16* tH     = (_Float16*)(ws + alloc((size_t)BB * HH * WW * CC * 2));  // 67 MB
  float*    scores = (float*)(ws + alloc((size_t)BB * HWA * 4));
  float*    boxes  = (float*)(ws + alloc((size_t)BB * HWA * 4 * 4));          // deltas -> boxes in place
  unsigned* hist   = (unsigned*)(ws + alloc((size_t)BB * 2048 * 4));
  unsigned* tcnt   = (unsigned*)(ws + alloc((size_t)BB * 4 * 4));
  unsigned* selKey = (unsigned*)(ws + alloc((size_t)BB * 2048 * 4));
  int*      selIdx = (int*)(ws + alloc((size_t)BB * 2048 * 4));
  float*    sbox   = (float*)(ws + alloc((size_t)BB * PRE * 4 * 4));
  float*    sscr   = (float*)(ws + alloc((size_t)BB * PRE * 4));
  unsigned* nmask  = (unsigned*)(ws + alloc((size_t)BB * PRE * NMSW * 4));
  (void)ws_size; (void)in_sizes; (void)n_in; (void)out_size;

  // stage 0: layout conversions (halo zero + coalesced transpose + weight swizzle)
  k_border<<<4128, 256, 0, stream>>>(featH);
  k_cvt_feat<<<16384, 256, 0, stream>>>(features, featH);
  k_pack_w3<<<2304, 256, 0, stream>>>(conv_w, wswz);
  k_pack_w1<<<80, 256, 0, stream>>>(obj_w, delta_w, w1swz);

  // stage 1: 3x3 conv + ReLU via WMMA implicit GEMM (32px x 32co per wave)
  k_conv3x3<<<4096, 256, 0, stream>>>(featH, wswz, conv_b, tH);

  // stage 2: fused 1x1 convs (obj + deltas) via WMMA
  k_conv1x1<<<5120, 256, 0, stream>>>(tH, w1swz, obj_b, delta_b, scores, boxes);

  // stage 3: decode deltas -> boxes (in place)
  k_decode<<<7680, 256, 0, stream>>>(boxes);

  // stage 4: per-image top-2000 via bucket select + bitonic sort
  k_zero<<<64, 256, 0, stream>>>(hist, BB * 2048);
  k_zero<<<64, 256, 0, stream>>>(selKey, BB * 2048);
  k_zero<<<64, 256, 0, stream>>>((unsigned*)selIdx, BB * 2048);
  k_hist<<<768, 256, 0, stream>>>(scores, hist);
  k_select<<<BB, 32, 0, stream>>>(hist, tcnt);
  k_compact<<<7680, 256, 0, stream>>>(scores, tcnt, selKey, selIdx);
  k_sortgather<<<BB, 1024, 0, stream>>>(selKey, selIdx, boxes, sbox, sscr);

  // stage 5: NMS + final top-1000 emit
  k_nmsmask<<<3938, 256, 0, stream>>>(sbox, nmask);
  k_nmsfinal<<<BB, 256, 0, stream>>>(nmask, sbox, sscr, out);
}
